// MultiHeadAttention_8358006358405
// MI455X (gfx1250) — matrix operand order repaired and checked by reading, not compiled, not run on hardware
//
#include <hip/hip_runtime.h>
#include <hip/hip_bf16.h>

typedef __bf16 bf16;
typedef __attribute__((ext_vector_type(4)))  __bf16 v4bf;
typedef __attribute__((ext_vector_type(8)))  __bf16 v8bf;
typedef __attribute__((ext_vector_type(16))) __bf16 v16bf;
typedef __attribute__((ext_vector_type(4)))  float  v4f;
typedef __attribute__((ext_vector_type(8)))  float  v8f;
typedef unsigned int u32;
typedef __attribute__((ext_vector_type(4))) unsigned int v4u;
typedef __attribute__((ext_vector_type(8))) int v8i;
typedef __attribute__((ext_vector_type(4))) int v4i;

#define DEV __device__ __forceinline__

#ifndef __has_builtin
#define __has_builtin(x) 0
#endif
#if __has_builtin(__builtin_amdgcn_tensor_load_to_lds) && \
    __has_builtin(__builtin_amdgcn_s_wait_tensorcnt)
#define HAVE_TDM 1
#else
#define HAVE_TDM 0
#endif

constexpr int Bd = 4;
constexpr int Td = 2048;
constexpr int Cd = 2048;
constexpr int Hd = 16;
constexpr int Dh = 128;
constexpr int Md = Bd * Td;   // 8192 rows
constexpr int N3 = 3 * Cd;    // 6144 qkv cols

// LDS tile: 128 rows x 32 K bf16, padded to pitch 40 elements (80 B) to spread banks
constexpr int LPITCH = 40;
constexpr int TE = 128 * LPITCH;   // elements per tile buffer

// ---------- WMMA fragment helpers (layouts per CDNA5 ISA §7.12.2) ----------

// A-matrix 16x32 bf16: lane<16 -> row=lane, K {0..7}(v0-3) {16..23}(v4-7);
//                      lane>=16 -> row=lane-16, K {8..15} {24..31}
DEV v16bf load_frag_a(const bf16* base, size_t stride, int lane) {
  const int row  = lane & 15;
  const int koff = (lane >> 4) << 3;     // 0 or 8
  const bf16* p = base + (size_t)row * stride + koff;
  v8bf lo = *(const v8bf*)(p);
  v8bf hi = *(const v8bf*)(p + 16);
  return __builtin_shufflevector(lo, hi, 0, 1, 2, 3, 4, 5, 6, 7,
                                          8, 9, 10, 11, 12, 13, 14, 15);
}

// B-matrix 32x16 bf16: lane<16 -> col=lane, K {0..7} {16..23}; lane>=16 -> col=lane-16, K {8..15} {24..31}.
// `base` points at (K=0,N=0) of N-major storage with K contiguous, pitch `stride`.
DEV v16bf load_frag_b(const bf16* base, size_t stride, int lane) {
  const int col  = lane & 15;
  const int koff = (lane >> 4) << 3;     // 0 or 8
  const bf16* p = base + (size_t)col * stride + koff;
  v8bf lo = *(const v8bf*)(p);
  v8bf hi = *(const v8bf*)(p + 16);
  return __builtin_shufflevector(lo, hi, 0, 1, 2, 3, 4, 5, 6, 7,
                                          8, 9, 10, 11, 12, 13, 14, 15);
}

DEV v8f wmma_bf16(v16bf a, v16bf b, v8f c) {
  return __builtin_amdgcn_wmma_f32_16x16x32_bf16(
      false, a, false, b, (short)0, c, false, false);
}

// ---------- Tensor Data Mover helpers ----------
#if HAVE_TDM
typedef __attribute__((address_space(3))) char lds_char;

DEV u32 lds_off_u32(const void* p) {
  return (u32)(size_t)(lds_char*)p;
}

// One TDM DMA: 32(K-contiguous) x 128(rows) bf16 tile, row stride = kstride
// elements, into LDS at byte offset lds_byte with 16B pad every 64B (pitch 80B).
// D# layout per CDNA5 ISA.
DEV void tdm_load_tile(u32 lds_byte, const bf16* g, u32 tensor_rows, u32 kstride) {
  const unsigned long long ga = (unsigned long long)(size_t)g;
  v4u g0;
  g0.x = 1u;                                              // count=1, user mode
  g0.y = lds_byte;                                        // lds_addr
  g0.z = (u32)ga;                                         // global_addr[31:0]
  g0.w = ((u32)(ga >> 32) & 0x01FFFFFFu) | 0x80000000u;   // addr[56:32] | type=2
  v8i g1;
  g1.s0 = (int)((1u << 16)     // data_size = 2 bytes
              | (1u << 20)     // pad_enable
              | (3u << 22)     // pad_interval = 16 DWORDs (64B)
              | (3u << 25));   // pad_amount   = 4 DWORDs (16B)
  g1.s1 = (int)((kstride & 0xFFFFu) << 16);               // tensor_dim0[15:0]
  g1.s2 = (int)((kstride >> 16) | ((tensor_rows & 0xFFFFu) << 16)); // dim0 hi | dim1 lo
  g1.s3 = (int)((tensor_rows >> 16) | (32u << 16));       // dim1 hi | tile_dim0=32
  g1.s4 = 128;                                            // tile_dim1=128, tile_dim2=0
  g1.s5 = (int)kstride;                                   // tensor_dim0_stride[31:0]
  g1.s6 = 0;
  g1.s7 = 0;
  const v4i z4 = {0, 0, 0, 0};
  const v8i z8 = {0, 0, 0, 0, 0, 0, 0, 0};
  __builtin_amdgcn_tensor_load_to_lds(g0, g1, z4, z4, z8, 0);
}
#endif

// ---------- shared GEMM main loop: C[128x128] block, 8 waves of 32x64 ----------
DEV void gemm_core(const bf16* __restrict__ A, const bf16* __restrict__ Bt,
                   int mb, int nb, u32 rowsA, u32 rowsB,
                   bf16* smem, int wid, int lane, int wm, int wn,
                   v8f acc[2][4]) {
#if HAVE_TDM
  const u32 lds_base = lds_off_u32(smem);
  if (wid == 0) {   // one wave drives the TDM; EXEC is ignored by tensor ops
    tdm_load_tile(lds_base,              A  + (size_t)mb * Cd, rowsA, Cd);
    tdm_load_tile(lds_base + 2 * TE * 2, Bt + (size_t)nb * Cd, rowsB, Cd);
  }
  for (int it = 0; it < Cd / 32; ++it) {
    const int cur = it & 1;
    if (wid == 0) {
      if (it + 1 < Cd / 32) {
        const int nxt = cur ^ 1;
        const int k0n = (it + 1) * 32;
        tdm_load_tile(lds_base + (u32)nxt * TE * 2,
                      A + (size_t)mb * Cd + k0n, rowsA, Cd);
        tdm_load_tile(lds_base + (u32)(2 + nxt) * TE * 2,
                      Bt + (size_t)nb * Cd + k0n, rowsB, Cd);
        __builtin_amdgcn_s_wait_tensorcnt(2);  // the 2 just-issued may remain
      } else {
        __builtin_amdgcn_s_wait_tensorcnt(0);
      }
    }
    __syncthreads();                            // cur buffers fully in LDS
    const bf16* sa = smem + cur * TE;
    const bf16* sb = smem + (2 + cur) * TE;
    v16bf a0 = load_frag_a(sa + (size_t)wm * LPITCH, LPITCH, lane);
    v16bf a1 = load_frag_a(sa + (size_t)(wm + 16) * LPITCH, LPITCH, lane);
#pragma unroll
    for (int j = 0; j < 4; ++j) {
      v16bf bfr = load_frag_b(sb + (size_t)(wn + j * 16) * LPITCH, LPITCH, lane);
      acc[0][j] = wmma_bf16(a0, bfr, acc[0][j]);
      acc[1][j] = wmma_bf16(a1, bfr, acc[1][j]);
    }
    __syncthreads();                            // done reading before overwrite
  }
#else
  // synchronous LDS staging fallback (also what the host pass parses)
  for (int it = 0; it < Cd / 32; ++it) {
    const int k0 = it * 32;
    const int tid = wid * 32 + lane;
    for (int c = tid; c < 512; c += 256) {
      const int row = c >> 2, k8 = (c & 3) * 8;
      *(v8bf*)(smem + row * LPITCH + k8) =
          *(const v8bf*)(A + (size_t)(mb + row) * Cd + k0 + k8);
      *(v8bf*)(smem + 2 * TE + row * LPITCH + k8) =
          *(const v8bf*)(Bt + (size_t)(nb + row) * Cd + k0 + k8);
    }
    __syncthreads();
    const bf16* sa = smem;
    const bf16* sb = smem + 2 * TE;
    v16bf a0 = load_frag_a(sa + (size_t)wm * LPITCH, LPITCH, lane);
    v16bf a1 = load_frag_a(sa + (size_t)(wm + 16) * LPITCH, LPITCH, lane);
#pragma unroll
    for (int j = 0; j < 4; ++j) {
      v16bf bfr = load_frag_b(sb + (size_t)(wn + j * 16) * LPITCH, LPITCH, lane);
      acc[0][j] = wmma_bf16(a0, bfr, acc[0][j]);
      acc[1][j] = wmma_bf16(a1, bfr, acc[1][j]);
    }
    __syncthreads();
  }
#endif
}

// ---------- elementwise fp32 -> bf16 ----------
__global__ __launch_bounds__(256) void k_convert(const float* __restrict__ s,
                                                 bf16* __restrict__ d, int n4) {
  int i = blockIdx.x * blockDim.x + threadIdx.x;
  if (i < n4) {
    v4f v = ((const v4f*)s)[i];
    v4bf o = {(bf16)v[0], (bf16)v[1], (bf16)v[2], (bf16)v[3]};
    ((v4bf*)d)[i] = o;
  }
}

// ---------- tiled transpose + convert: src (Kd x Nd) f32 -> dst (Nd x Kd) bf16 ----------
__global__ __launch_bounds__(256) void k_transpose(const float* __restrict__ s,
                                                   bf16* __restrict__ d,
                                                   int Kd, int Nd) {
  __shared__ float tile[32][33];
  const int nb = blockIdx.x * 32;
  const int kb = blockIdx.y * 32;
  const int tx = threadIdx.x, ty = threadIdx.y;
#pragma unroll
  for (int i = 0; i < 32; i += 8)
    tile[ty + i][tx] = s[(size_t)(kb + ty + i) * Nd + nb + tx];
  __syncthreads();
#pragma unroll
  for (int i = 0; i < 32; i += 8)
    d[(size_t)(nb + ty + i) * Kd + kb + tx] = (bf16)tile[tx][ty + i];
}

// ---------- QKV GEMM + bias + fused RoPE + head scatter ----------
// A: (M,K) bf16; Bt: (N,K) bf16. q,k -> (B,H,T,Dh); v -> (B,H,Dh,T) transposed.
__global__ __launch_bounds__(256) void k_gemm_qkv_rope(
    const bf16* __restrict__ A, const bf16* __restrict__ Bt,
    const float* __restrict__ bias,
    bf16* __restrict__ qo, bf16* __restrict__ ko, bf16* __restrict__ vto) {
  __shared__ __align__(16) bf16 smem[4 * TE];
  const int lane = threadIdx.x & 31;
  const int wid  = threadIdx.x >> 5;
  const int hi   = lane >> 4;
  const int ln   = lane & 15;
  const int wm = (wid >> 1) * 32;
  const int wn = (wid & 1) * 64;
  const int mb = blockIdx.y * 128;
  const int nb = blockIdx.x * 128;
  const int m0 = mb + wm;
  const int n0 = nb + wn;

  v8f acc[2][4] = {};
  gemm_core(A, Bt, mb, nb, (u32)Md, (u32)N3, smem, wid, lane, wm, wn, acc);

  const float kinv = 9.210340371976184f / (float)Cd;   // ln(10000)/C
#pragma unroll
  for (int i = 0; i < 2; ++i) {
#pragma unroll
    for (int j = 0; j < 4; ++j) {
      const int col = n0 + j * 16 + ln;          // global qkv column
      const float bv = bias[col];
      const bool isqk = col < 2 * Cd;            // uniform across the 16-col frag
      const int cidx = isqk ? (col < Cd ? col : col - Cd) : (col - 2 * Cd);
      const int h = cidx >> 7;
      const int d = cidx & (Dh - 1);
      const int odd = cidx & 1;
      const float freq = __expf(-(float)(cidx & ~1) * kinv);
#pragma unroll
      for (int r = 0; r < 8; ++r) {
        const int m  = m0 + i * 16 + r + 8 * hi; // C-frag row
        const int bb = m >> 11;                  // batch
        const int t  = m & (Td - 1);             // token
        float val = acc[i][j][r] + bv;
        if (isqk) {
          // pair channel (c^1) is the adjacent lane of this C-fragment
          const float other = __shfl_xor(val, 1, 32);
          float sn, cs;
          __sincosf((float)t * freq, &sn, &cs);
          const float x0 = odd ? other : val;
          const float x1 = odd ? val : other;
          const float out = odd ? (x0 * sn + x1 * cs) : (x0 * cs - x1 * sn);
          bf16* dst = (col < Cd) ? qo : ko;
          dst[(((size_t)(bb * Hd + h) * Td + t) << 7) + d] = (bf16)out;
        } else {
          vto[((size_t)(bb * Hd + h) * Dh + d) * Td + t] = (bf16)val;
        }
      }
    }
  }
}

// ---------- flash attention: one wave per (b,h, 16-query tile) ----------
__global__ __launch_bounds__(256) void k_attn(const bf16* __restrict__ q,
                                              const bf16* __restrict__ k,
                                              const bf16* __restrict__ vt,
                                              bf16* __restrict__ y) {
  __shared__ __align__(16) bf16 plds[8][16 * 40];   // per-wave 16x32 P scratch
  const int lane = threadIdx.x & 31;
  const int wid  = threadIdx.x >> 5;
  const int hi   = lane >> 4;
  const int ln   = lane & 15;
  const int gw = blockIdx.x * 8 + wid;
  const int bh = gw >> 7;
  const int qt = gw & 127;
  const int bb = bh >> 4;
  const int h  = bh & 15;
  const bf16* qb = q  + (size_t)bh * Td * Dh;
  const bf16* kb = k  + (size_t)bh * Td * Dh;
  const bf16* vb = vt + (size_t)bh * Dh * Td;
  const int q0 = qt * 16;

  v16bf qf[4];
#pragma unroll
  for (int kk = 0; kk < 4; ++kk)
    qf[kk] = load_frag_a(qb + (size_t)q0 * Dh + kk * 32, Dh, lane);

  v8f o[8] = {};
  float mrow[8], lrow[8];
#pragma unroll
  for (int r = 0; r < 8; ++r) { mrow[r] = -3.0e38f; lrow[r] = 0.f; }

  const float scale = 0.08838834764831845f;   // 1/sqrt(128)
  const int ntile = ((q0 + 15) >> 5) + 1;     // causal: 32-key tiles
  for (int it = 0; it < ntile; ++it) {
    const int kb0 = it * 32;
    v8f s0 = {}, s1 = {};
#pragma unroll
    for (int kk = 0; kk < 4; ++kk) {
      s0 = wmma_bf16(qf[kk], load_frag_b(kb + (size_t)kb0 * Dh + kk * 32, Dh, lane), s0);
      s1 = wmma_bf16(qf[kk], load_frag_b(kb + (size_t)(kb0 + 16) * Dh + kk * 32, Dh, lane), s1);
    }
#pragma unroll
    for (int r = 0; r < 8; ++r) {
      const int qrow = q0 + r + 8 * hi;
      float v0 = s0[r] * scale;
      float v1 = s1[r] * scale;
      if (kb0 + ln > qrow)      v0 = -3.0e38f;
      if (kb0 + 16 + ln > qrow) v1 = -3.0e38f;
      float mx = fmaxf(v0, v1);
#pragma unroll
      for (int dlt = 1; dlt < 16; dlt <<= 1) mx = fmaxf(mx, __shfl_xor(mx, dlt, 32));
      const float mnew = fmaxf(mrow[r], mx);
      const float p0 = __expf(v0 - mnew);
      const float p1 = __expf(v1 - mnew);
      float sum = p0 + p1;
#pragma unroll
      for (int dlt = 1; dlt < 16; dlt <<= 1) sum += __shfl_xor(sum, dlt, 32);
      const float alpha = __expf(mrow[r] - mnew);
      lrow[r] = lrow[r] * alpha + sum;
      mrow[r] = mnew;
#pragma unroll
      for (int c = 0; c < 8; ++c) o[c][r] *= alpha;
      const int prow = r + 8 * hi;
      plds[wid][prow * 40 + ln]      = (bf16)p0;
      plds[wid][prow * 40 + 16 + ln] = (bf16)p1;
    }
    asm volatile("" ::: "memory");   // wave32 DS pipe is in-order; just pin order
    const v16bf pa = load_frag_a(&plds[wid][0], 40, lane);
#pragma unroll
    for (int c = 0; c < 8; ++c)
      o[c] = wmma_bf16(pa, load_frag_b(vb + (size_t)(c * 16) * Td + kb0, Td, lane), o[c]);
  }

#pragma unroll
  for (int c = 0; c < 8; ++c) {
    const int d = c * 16 + ln;
#pragma unroll
    for (int r = 0; r < 8; ++r) {
      const int qrow = q0 + r + 8 * hi;
      const float val = o[c][r] / lrow[r];
      y[(size_t)(bb * Td + qrow) * Cd + h * Dh + d] = (bf16)val;
    }
  }
}

// ---------- output projection GEMM + bias -> fp32 ----------
__global__ __launch_bounds__(256) void k_gemm_proj(const bf16* __restrict__ A,
                                                   const bf16* __restrict__ Bt,
                                                   const float* __restrict__ bias,
                                                   float* __restrict__ out) {
  __shared__ __align__(16) bf16 smem[4 * TE];
  const int lane = threadIdx.x & 31;
  const int wid  = threadIdx.x >> 5;
  const int hi   = lane >> 4;
  const int ln   = lane & 15;
  const int wm = (wid >> 1) * 32;
  const int wn = (wid & 1) * 64;
  const int mb = blockIdx.y * 128;
  const int nb = blockIdx.x * 128;

  v8f acc[2][4] = {};
  gemm_core(A, Bt, mb, nb, (u32)Md, (u32)Cd, smem, wid, lane, wm, wn, acc);

#pragma unroll
  for (int i = 0; i < 2; ++i) {
#pragma unroll
    for (int j = 0; j < 4; ++j) {
      const int col = nb + wn + j * 16 + ln;
      const float bv = bias[col];
#pragma unroll
      for (int r = 0; r < 8; ++r) {
        const int m = mb + wm + i * 16 + r + 8 * hi;
        out[(size_t)m * Cd + col] = acc[i][j][r] + bv;
      }
    }
  }
}

extern "C" void kernel_launch(void* const* d_in, const int* in_sizes, int n_in,
                              void* d_out, int out_size, void* d_ws, size_t ws_size,
                              hipStream_t stream) {
  const float* x      = (const float*)d_in[0];
  const float* w_qkv  = (const float*)d_in[1];
  const float* b_qkv  = (const float*)d_in[2];
  const float* w_proj = (const float*)d_in[3];
  const float* b_proj = (const float*)d_in[4];
  float* out = (float*)d_out;

  char* ws = (char*)d_ws;
  bf16* xb     = (bf16*)ws; ws += (size_t)Md * Cd * sizeof(bf16);
  bf16* wqkvT  = (bf16*)ws; ws += (size_t)N3 * Cd * sizeof(bf16);
  bf16* wprojT = (bf16*)ws; ws += (size_t)Cd * Cd * sizeof(bf16);
  bf16* qbuf   = (bf16*)ws; ws += (size_t)Md * Cd * sizeof(bf16);
  bf16* kbuf   = (bf16*)ws; ws += (size_t)Md * Cd * sizeof(bf16);
  bf16* vtbuf  = (bf16*)ws; ws += (size_t)Md * Cd * sizeof(bf16);
  bf16* ybuf   = (bf16*)ws;

  const int n4 = Md * Cd / 4;
  k_convert<<<(n4 + 255) / 256, 256, 0, stream>>>(x, xb, n4);

  dim3 tb(32, 8);
  k_transpose<<<dim3(N3 / 32, Cd / 32), tb, 0, stream>>>(w_qkv, wqkvT, Cd, N3);
  k_transpose<<<dim3(Cd / 32, Cd / 32), tb, 0, stream>>>(w_proj, wprojT, Cd, Cd);

  k_gemm_qkv_rope<<<dim3(N3 / 128, Md / 128), 256, 0, stream>>>(
      xb, wqkvT, b_qkv, qbuf, kbuf, vtbuf);

  k_attn<<<(Bd * Hd * (Td / 16)) / 8, 256, 0, stream>>>(qbuf, kbuf, vtbuf, ybuf);

  k_gemm_proj<<<dim3(Cd / 128, Md / 128), 256, 0, stream>>>(ybuf, wprojT, b_proj, out);
}
